// RTDETRDecoderCore_52596169506902
// MI455X (gfx1250) — matrix operand order repaired and checked by reading, not compiled, not run on hardware
//
#include <hip/hip_runtime.h>

typedef __attribute__((ext_vector_type(16))) __bf16 v16bf;
typedef __attribute__((ext_vector_type(8)))  float  v8f;

#define HD     256
#define T_TOK  8400
#define NQ     300
#define NC     80
#define BS     32
#define KSTEPS 8      // 256 / 32
#define XSTR   260    // LDS row stride (floats), padded to dodge bank conflicts

// ---------- helpers ----------

__device__ __forceinline__ __bf16 f2bf(float f) {
  unsigned u = __float_as_uint(f);
  unsigned lsb = (u >> 16) & 1u;
  u += 0x7FFFu + lsb;                       // round-to-nearest-even
  unsigned short h = (unsigned short)(u >> 16);
  return __builtin_bit_cast(__bf16, h);
}

__device__ __forceinline__ void token_geom(int t, int& gx, int& gy, int& w, int& h, float& wh) {
  int idx;
  if (t < 6400)      { idx = t;        w = 80; h = 80; wh = 0.05f; }
  else if (t < 8000) { idx = t - 6400; w = 40; h = 40; wh = 0.10f; }
  else               { idx = t - 8000; w = 20; h = 20; wh = 0.20f; }
  gy = idx / w; gx = idx % w;
}

__device__ __forceinline__ bool token_valid(int t) {
  int gx, gy, w, h; float wh;
  token_geom(t, gx, gy, w, h, wh);
  float x = (gx + 0.5f) / (float)w, y = (gy + 0.5f) / (float)h;
  const float e = 0.01f;
  return (x > e) && (x < 1.f - e) && (y > e) && (y < 1.f - e) &&
         (wh > e) && (wh < 1.f - e);
}

__device__ __forceinline__ void anchor4(int t, float a[4]) {
  int gx, gy, w, h; float wh;
  token_geom(t, gx, gy, w, h, wh);
  float v[4];
  v[0] = (gx + 0.5f) / (float)w;
  v[1] = (gy + 0.5f) / (float)h;
  v[2] = wh; v[3] = wh;
  const float e = 0.01f;
  bool valid = true;
  #pragma unroll
  for (int i = 0; i < 4; i++) valid = valid && (v[i] > e) && (v[i] < 1.f - e);
  #pragma unroll
  for (int i = 0; i < 4; i++) a[i] = valid ? logf(v[i] / (1.f - v[i])) : INFINITY;
}

// A-matrix 16x32 bf16 fragment from an LDS f32 row (ISA 7.12.2 layout):
// lanes 0-15: elems 0..7 -> K 0..7,  elems 8..15 -> K 16..23
// lanes16-31: elems 0..7 -> K 8..15, elems 8..15 -> K 24..31
__device__ __forceinline__ v16bf load_afrag(const float* xrow, int kstep) {
  int lane = threadIdx.x & 31;
  int kb0 = kstep * 32 + ((lane >= 16) ? 8 : 0);
  int kb1 = kstep * 32 + ((lane >= 16) ? 24 : 16);
  v16bf a;
  #pragma unroll
  for (int j = 0; j < 8; j++) a[j] = f2bf(xrow[kb0 + j]);
  #pragma unroll
  for (int j = 0; j < 8; j++) a[8 + j] = f2bf(xrow[kb1 + j]);
  return a;
}

// B fragments are prepacked: 16 contiguous bf16 per lane (32B coalesced load).
__device__ __forceinline__ v16bf load_bfrag(const __bf16* pack, int ntiles, int kstep, int ntile) {
  int lane = threadIdx.x & 31;
  const __bf16* p = pack + ((((size_t)kstep * ntiles + ntile) * 32 + lane) << 4);
  return *(const v16bf*)p;
}

// Block-wide GEMM: D[16 x ntiles*16] = A[16x256] * Bpack + bias, optional ReLU.
// A read from LDS (stride XSTR), D written to LDS (stride dstride).
__device__ __forceinline__ void block_gemm_bf16(const float* a_lds, const __bf16* pack,
                                                const float* bias, float* d_lds,
                                                int dstride, int ntiles, bool relu) {
  int lane = threadIdx.x & 31;
  int wave = threadIdx.x >> 5;
  int m = lane & 15;
  const float* arow = a_lds + m * XSTR;
  v16bf afr[KSTEPS];
  #pragma unroll
  for (int ks = 0; ks < KSTEPS; ks++) afr[ks] = load_afrag(arow, ks);

  for (int nt = wave; nt < ntiles; nt += 8) {
    v8f acc = {0.f, 0.f, 0.f, 0.f, 0.f, 0.f, 0.f, 0.f};
    #pragma unroll
    for (int ks = 0; ks < KSTEPS; ks++) {
      v16bf bf = load_bfrag(pack, ntiles, ks, nt);
      acc = __builtin_amdgcn_wmma_f32_16x16x32_bf16(
          /*neg_a=*/false, afr[ks], /*neg_b=*/false, bf,
          /*c_mod=*/(short)0, acc, /*reuse_a=*/false, /*reuse_b=*/false);
    }
    int col = nt * 16 + (lane & 15);
    float bv = bias[col];
    int rbase = (lane >= 16) ? 8 : 0;
    #pragma unroll
    for (int r = 0; r < 8; r++) {
      float v = acc[r] + bv;
      if (relu) v = fmaxf(v, 0.f);
      d_lds[(rbase + r) * dstride + col] = v;
    }
  }
}

// ---------- kernel 0: weight prepack f32 -> bf16 B-fragment layout ----------
__global__ __launch_bounds__(256) void k0_prepack(const float* __restrict__ W,
                                                  __bf16* __restrict__ pack,
                                                  int N, int ntiles) {
  int i = blockIdx.x * 256 + threadIdx.x;
  int total = KSTEPS * ntiles * 512;
  if (i >= total) return;
  int j    = i & 15;
  int lane = (i >> 4) & 31;
  int tile = i >> 9;
  int nt = tile % ntiles;
  int ks = tile / ntiles;
  int k = ks * 32 + ((lane >= 16) ? 8 : 0) + (j & 7) + ((j >= 8) ? 16 : 0);   // B layout: K = half*8 + elem (elem<8), 16 + half*8 + elem-8 (elem>=8)
  int n = nt * 16 + (lane & 15);
  float v = (n < N) ? W[(size_t)k * N + n] : 0.f;
  pack[i] = f2bf(v);
}

// ---------- kernel 1: features GEMM + LN + scores + per-token max ----------
__global__ __launch_bounds__(256) void k1_feat_score(
    const float* __restrict__ feats, const __bf16* __restrict__ encw_p,
    const float* __restrict__ enc_b, const float* __restrict__ ln_g,
    const float* __restrict__ ln_b, const __bf16* __restrict__ scw_p,
    const float* __restrict__ score_b, float* __restrict__ maxscore) {
  __shared__ float xs[16 * XSTR];
  __shared__ float hs[16 * XSTR];
  __shared__ float sc[16][NC];
  __shared__ float mu_s[16], rs_s[16];

  int b = blockIdx.y;
  int t0 = blockIdx.x * 16;
  int tid = threadIdx.x;

  { // masked, coalesced load of 16 tokens x 256
    int r = tid >> 4, seg = tid & 15;
    int t = t0 + r;
    float vm = token_valid(t) ? 1.f : 0.f;
    const float* src = feats + ((size_t)b * T_TOK + t) * HD + seg * 16;
    float* dst = xs + r * XSTR + seg * 16;
    #pragma unroll
    for (int j = 0; j < 16; j++) dst[j] = vm * src[j];
  }
  __syncthreads();

  block_gemm_bf16(xs, encw_p, enc_b, hs, XSTR, 16, false);
  __syncthreads();

  if (tid < 16) { // LN stats
    float s = 0.f, q = 0.f;
    const float* hrow = hs + tid * XSTR;
    for (int c = 0; c < HD; c++) { float v = hrow[c]; s += v; q += v * v; }
    float mu = s * (1.f / HD);
    float var = q * (1.f / HD) - mu * mu;
    mu_s[tid] = mu;
    rs_s[tid] = rsqrtf(var + 1e-5f);
  }
  __syncthreads();
  { // normalize in place
    #pragma unroll
    for (int j = 0; j < 16; j++) {
      int e = tid * 16 + j;
      int r = e >> 8, c = e & 255;
      float v = hs[r * XSTR + c];
      hs[r * XSTR + c] = (v - mu_s[r]) * rs_s[r] * ln_g[c] + ln_b[c];
    }
  }
  __syncthreads();

  block_gemm_bf16(hs, scw_p, score_b, &sc[0][0], NC, 5, false);
  __syncthreads();

  if (tid < 16) {
    float mx = -INFINITY;
    for (int c = 0; c < NC; c++) mx = fmaxf(mx, sc[tid][c]);
    maxscore[(size_t)b * T_TOK + t0 + tid] = mx;
  }
}

// ---------- kernel 2: exact top-300 per batch (lax.top_k order) ----------
__global__ __launch_bounds__(256) void k2_topk(const float* __restrict__ maxscore,
                                               int* __restrict__ topk) {
  __shared__ float s[T_TOK];
  __shared__ float rv[256];
  __shared__ int   ri[256];
  int b = blockIdx.x, tid = threadIdx.x;
  for (int i = tid; i < T_TOK; i += 256) s[i] = maxscore[(size_t)b * T_TOK + i];
  __syncthreads();
  for (int q = 0; q < NQ; q++) {
    float bv = -INFINITY; int bi = T_TOK;
    for (int i = tid; i < T_TOK; i += 256) {
      float v = s[i];
      if (v > bv || (v == bv && i < bi)) { bv = v; bi = i; }
    }
    rv[tid] = bv; ri[tid] = bi;
    __syncthreads();
    for (int off = 128; off > 0; off >>= 1) {
      if (tid < off) {
        float ov = rv[tid + off]; int oi = ri[tid + off];
        if (ov > rv[tid] || (ov == rv[tid] && oi < ri[tid])) { rv[tid] = ov; ri[tid] = oi; }
      }
      __syncthreads();
    }
    if (tid == 0) { topk[b * NQ + q] = ri[0]; s[ri[0]] = -INFINITY; }
    __syncthreads();
  }
}

// ---------- kernel 3: gather + recompute features + scores + bbox MLP ----------
__global__ __launch_bounds__(256) void k3_decode(
    const float* __restrict__ feats, const __bf16* __restrict__ encw_p,
    const float* __restrict__ enc_b, const float* __restrict__ ln_g,
    const float* __restrict__ ln_b, const __bf16* __restrict__ scw_p,
    const float* __restrict__ score_b, const __bf16* __restrict__ w1_p,
    const float* __restrict__ bb_b1, const __bf16* __restrict__ w2_p,
    const float* __restrict__ bb_b2, const float* __restrict__ bb_w3,
    const float* __restrict__ bb_b3, const int* __restrict__ topk,
    float* __restrict__ out_emb, float* __restrict__ out_rb,
    float* __restrict__ out_bb, float* __restrict__ out_sc) {
  __shared__ float xs[16 * XSTR];   // gathered x, later m1
  __shared__ float fs[16 * XSTR];   // features, later m2
  __shared__ float sc[16][NC];
  __shared__ float mu_s[16], rs_s[16];
  __shared__ int   tok_s[16];
  __shared__ float anc_s[16][4];
  __shared__ float delta_s[16][4];

  int b = blockIdx.y;
  int q0 = blockIdx.x * 16;
  int tid = threadIdx.x;

  if (tid < 16) {
    int q = q0 + tid; if (q > NQ - 1) q = NQ - 1;      // clamp partial tile
    tok_s[tid] = topk[b * NQ + q];
  }
  __syncthreads();

  { // gather masked feats rows
    int r = tid >> 4, seg = tid & 15;
    int t = tok_s[r];
    float vm = token_valid(t) ? 1.f : 0.f;
    const float* src = feats + ((size_t)b * T_TOK + t) * HD + seg * 16;
    float* dst = xs + r * XSTR + seg * 16;
    #pragma unroll
    for (int j = 0; j < 16; j++) dst[j] = vm * src[j];
  }
  __syncthreads();

  block_gemm_bf16(xs, encw_p, enc_b, fs, XSTR, 16, false);
  __syncthreads();

  if (tid < 16) {
    float s = 0.f, q = 0.f;
    const float* hrow = fs + tid * XSTR;
    for (int c = 0; c < HD; c++) { float v = hrow[c]; s += v; q += v * v; }
    float mu = s * (1.f / HD);
    float var = q * (1.f / HD) - mu * mu;
    mu_s[tid] = mu;
    rs_s[tid] = rsqrtf(var + 1e-5f);
  }
  __syncthreads();
  {
    #pragma unroll
    for (int j = 0; j < 16; j++) {
      int e = tid * 16 + j;
      int r = e >> 8, c = e & 255;
      float v = fs[r * XSTR + c];
      fs[r * XSTR + c] = (v - mu_s[r]) * rs_s[r] * ln_g[c] + ln_b[c];
    }
  }
  __syncthreads();

  { // embeddings out
    int r = tid >> 4, seg = tid & 15;
    int q = q0 + r;
    if (q < NQ) {
      float* dst = out_emb + ((size_t)b * NQ + q) * HD + seg * 16;
      #pragma unroll
      for (int j = 0; j < 16; j++) dst[j] = fs[r * XSTR + seg * 16 + j];
    }
  }

  block_gemm_bf16(fs, scw_p, score_b, &sc[0][0], NC, 5, false);
  __syncthreads();

  for (int i = tid; i < 16 * NC; i += 256) { // enc_scores out
    int r = i / NC, c = i % NC;
    int q = q0 + r;
    if (q < NQ) out_sc[((size_t)b * NQ + q) * NC + c] = sc[r][c];
  }

  block_gemm_bf16(fs, w1_p, bb_b1, xs, XSTR, 16, true);   // m1 -> xs
  __syncthreads();
  block_gemm_bf16(xs, w2_p, bb_b2, fs, XSTR, 16, true);   // m2 -> fs
  __syncthreads();

  if (tid < 64) { // bbox_delta = m2 @ bb_w3 + b3 (256 -> 4)
    int r = tid >> 2, c = tid & 3;
    float s = bb_b3[c];
    for (int k = 0; k < HD; k++) s += fs[r * XSTR + k] * bb_w3[k * 4 + c];
    delta_s[r][c] = s;
  }
  if (tid < 16) anchor4(tok_s[tid], anc_s[tid]);
  __syncthreads();

  if (tid < 64) {
    int r = tid >> 2, c = tid & 3;
    int q = q0 + r;
    if (q < NQ) {
      float rb = delta_s[r][c] + anc_s[r][c];
      size_t o = ((size_t)b * NQ + q) * 4 + c;
      out_rb[o] = rb;
      out_bb[o] = 1.f / (1.f + expf(-rb));
    }
  }
}

// ---------- launch ----------
extern "C" void kernel_launch(void* const* d_in, const int* in_sizes, int n_in,
                              void* d_out, int out_size, void* d_ws, size_t ws_size,
                              hipStream_t stream) {
  (void)in_sizes; (void)n_in; (void)out_size; (void)ws_size;
  const float* feats   = (const float*)d_in[0];
  const float* enc_w   = (const float*)d_in[1];
  const float* enc_b   = (const float*)d_in[2];
  const float* ln_g    = (const float*)d_in[3];
  const float* ln_b    = (const float*)d_in[4];
  const float* score_w = (const float*)d_in[5];
  const float* score_b = (const float*)d_in[6];
  const float* bb_w1   = (const float*)d_in[7];
  const float* bb_b1   = (const float*)d_in[8];
  const float* bb_w2   = (const float*)d_in[9];
  const float* bb_b2   = (const float*)d_in[10];
  const float* bb_w3   = (const float*)d_in[11];
  const float* bb_b3   = (const float*)d_in[12];

  char* ws = (char*)d_ws;
  __bf16* encw_p  = (__bf16*)(ws + 0);         // 8*16*512*2  = 131072 B
  __bf16* scw_p   = (__bf16*)(ws + 131072);    // 8*5*512*2   =  40960 B
  __bf16* w1_p    = (__bf16*)(ws + 172032);    // 131072 B
  __bf16* w2_p    = (__bf16*)(ws + 303104);    // 131072 B
  float*  maxsc   = (float*) (ws + 434176);    // 32*8400*4   = 1075200 B
  int*    topk    = (int*)   (ws + 1509376);   // 32*300*4    =  38400 B

  k0_prepack<<<dim3((KSTEPS * 16 * 512 + 255) / 256), 256, 0, stream>>>(enc_w, encw_p, HD, 16);
  k0_prepack<<<dim3((KSTEPS * 5  * 512 + 255) / 256), 256, 0, stream>>>(score_w, scw_p, NC, 5);
  k0_prepack<<<dim3((KSTEPS * 16 * 512 + 255) / 256), 256, 0, stream>>>(bb_w1, w1_p, HD, 16);
  k0_prepack<<<dim3((KSTEPS * 16 * 512 + 255) / 256), 256, 0, stream>>>(bb_w2, w2_p, HD, 16);

  k1_feat_score<<<dim3(T_TOK / 16, BS), 256, 0, stream>>>(
      feats, encw_p, enc_b, ln_g, ln_b, scw_p, score_b, maxsc);

  k2_topk<<<dim3(BS), 256, 0, stream>>>(maxsc, topk);

  float* out     = (float*)d_out;
  float* out_emb = out;
  float* out_rb  = out_emb + (size_t)BS * NQ * HD;
  float* out_bb  = out_rb  + (size_t)BS * NQ * 4;
  float* out_sc  = out_bb  + (size_t)BS * NQ * 4;

  k3_decode<<<dim3((NQ + 15) / 16, BS), 256, 0, stream>>>(
      feats, encw_p, enc_b, ln_g, ln_b, scw_p, score_b,
      w1_p, bb_b1, w2_p, bb_b2, bb_w3, bb_b3, topk,
      out_emb, out_rb, out_bb, out_sc);
}
